// PerceiverAttention_67525475828186
// MI455X (gfx1250) — matrix operand order repaired and checked by reading, not compiled, not run on hardware
//
#include <hip/hip_runtime.h>
#include <hip/hip_bf16.h>

typedef _Float16 v16h __attribute__((ext_vector_type(16)));
typedef _Float16 v8h  __attribute__((ext_vector_type(8)));
typedef float    v8f  __attribute__((ext_vector_type(8)));

#define BATCH 16
#define SEQ   64
#define MLAT  4096
#define DIM   1024
#define HEADS 16
#define DHEAD 64

__device__ __forceinline__ v8f wmma16(v16h a, v16h b, v8f c) {
    return __builtin_amdgcn_wmma_f32_16x16x32_f16(false, a, false, b, (short)0, c, false, false);
}

// A-operand fragment: row-major source, per-lane row = lane%16,
// halves 0..15 = K[kbase..kbase+15],
// kbase = kk + (lane<16 ? 0 : 16)
__device__ __forceinline__ v16h load_a(const _Float16* rowptr, int kbase) {
    v8h lo = *(const v8h*)(rowptr + kbase);
    v8h hi = *(const v8h*)(rowptr + kbase + 8);
    return __builtin_shufflevector(lo, hi, 0,1,2,3,4,5,6,7,8,9,10,11,12,13,14,15);
}

// ---------------- Kernel 0: weights -> transposed f16 ----------------
__global__ __launch_bounds__(256) void wcvt_kernel(const float* __restrict__ Wq,
                                                   const float* __restrict__ Wkv,
                                                   const float* __restrict__ Wout,
                                                   _Float16* __restrict__ WqT,
                                                   _Float16* __restrict__ WkvT,
                                                   _Float16* __restrict__ WoutT) {
    const int NQ = DIM * DIM;          // 1M
    const int NKV = DIM * 2 * DIM;     // 2M
    int id = blockIdx.x * 256 + threadIdx.x;
    if (id < NQ) {
        int n = id >> 10, k = id & 1023;
        WqT[id] = (_Float16)Wq[(size_t)k * DIM + n];
    } else if (id < NQ + NKV) {
        int i = id - NQ;
        int n = i >> 10, k = i & 1023;               // n in [0,2048)
        WkvT[i] = (_Float16)Wkv[(size_t)k * 2048 + n];
    } else if (id < NQ + NKV + NQ) {
        int i = id - NQ - NKV;
        int n = i >> 10, k = i & 1023;
        WoutT[i] = (_Float16)Wout[(size_t)k * DIM + n];
    }
}

// ---------------- Kernel 1: LN(x) + K/V projection ----------------
// grid = BATCH * (SEQ/16) = 64 blocks, 256 threads (8 waves)
__global__ __launch_bounds__(256) void kv_kernel(const float* __restrict__ x,
                                                 const float* __restrict__ gx,
                                                 const float* __restrict__ bx,
                                                 const _Float16* __restrict__ WkvT,
                                                 _Float16* __restrict__ K16,
                                                 _Float16* __restrict__ V16) {
    __shared__ _Float16 xt[16][DIM];   // 32 KB
    const int b  = blockIdx.x >> 2;
    const int s0 = (blockIdx.x & 3) * 16;
    const int wave = threadIdx.x >> 5, lane = threadIdx.x & 31;

    // ---- LayerNorm of 16 rows (2 rows per wave) ----
    for (int rr = 0; rr < 2; ++rr) {
        int row = wave * 2 + rr;
        const float* xp = x + ((size_t)b * SEQ + s0 + row) * DIM;
        float xv[32]; float s1 = 0.f, s2 = 0.f;
        #pragma unroll
        for (int i = 0; i < 32; ++i) { float v = xp[lane + 32 * i]; xv[i] = v; s1 += v; s2 += v * v; }
        for (int off = 16; off > 0; off >>= 1) { s1 += __shfl_xor(s1, off, 32); s2 += __shfl_xor(s2, off, 32); }
        float mean = s1 * (1.f / DIM);
        float rstd = rsqrtf(s2 * (1.f / DIM) - mean * mean + 1e-5f);
        #pragma unroll
        for (int i = 0; i < 32; ++i) {
            int c = lane + 32 * i;
            xt[row][c] = (_Float16)((xv[i] - mean) * rstd * gx[c] + bx[c]);
        }
    }
    __syncthreads();

    // ---- GEMM: (16 x 1024) @ WkvT' -> 2048 outputs; 128 n-tiles, 16 per wave ----
    const int arow  = lane & 15;
    const int koffA = (lane < 16) ? 0 : 16;
    const int koffB = (lane < 16) ? 0 : 16;
    for (int t = 0; t < 16; ++t) {
        int n0 = (wave + 8 * t) * 16;
        int nrow = n0 + (lane & 15);
        v8f acc = {};
        const _Float16* wb = WkvT + (size_t)nrow * DIM;
        for (int kk = 0; kk < DIM; kk += 32) {
            v16h A = load_a(&xt[arow][0], kk + koffA);
            v16h B = *(const v16h*)(wb + kk + koffB);
            acc = wmma16(A, B, acc);
        }
        #pragma unroll
        for (int r = 0; r < 8; ++r) {
            int srow = s0 + r + ((lane < 16) ? 0 : 8);
            float val = acc[r];
            if (nrow < DIM) {
                int h = nrow >> 6, d = nrow & 63;
                K16[(((size_t)b * HEADS + h) * SEQ + srow) * DHEAD + d] = (_Float16)val;
            } else {
                int n2 = nrow - DIM; int h = n2 >> 6, d = n2 & 63;
                V16[(((size_t)b * HEADS + h) * SEQ + srow) * DHEAD + d] = (_Float16)val;
            }
        }
    }
}

// ---------------- Kernel 2: fused LN(latents) + Q + attention + out-proj ----------------
// grid = BATCH * (MLAT/32) = 2048 blocks, 256 threads (8 waves)
// dynamic LDS: lat(64KB) | q16(4KB) | simS(8KB) | at16(4KB) | ao16(4KB) = 84KB
#define ATTN_LDS_BYTES (32 * DIM * 2 + 32 * 64 * 2 + 32 * 64 * 4 + 32 * 64 * 2 + 32 * 64 * 2)

__global__ __launch_bounds__(256) void attn_kernel(const float* __restrict__ latents,
                                                   const float* __restrict__ gl,
                                                   const float* __restrict__ bl,
                                                   const _Float16* __restrict__ WqT,
                                                   const _Float16* __restrict__ WoutT,
                                                   const _Float16* __restrict__ K16,
                                                   const _Float16* __restrict__ V16,
                                                   float* __restrict__ out) {
    extern __shared__ char smem[];
    _Float16* lat  = (_Float16*)smem;                 // [32][1024]
    _Float16* q16  = lat + 32 * DIM;                  // [32][64]
    float*    simS = (float*)(q16 + 32 * 64);         // [32][64]
    _Float16* at16 = (_Float16*)(simS + 32 * 64);     // [32][64]
    _Float16* ao16 = at16 + 32 * 64;                  // [32][64]

    const int b  = blockIdx.x >> 7;           // MLAT/32 = 128 tiles per batch
    const int m0 = (blockIdx.x & 127) * 32;
    const int wave = threadIdx.x >> 5, lane = threadIdx.x & 31;

    // ---- LayerNorm of 32 latent rows (4 rows per wave) ----
    for (int rr = 0; rr < 4; ++rr) {
        int row = wave * 4 + rr;
        const float* lp = latents + ((size_t)b * MLAT + m0 + row) * DIM;
        float xv[32]; float s1 = 0.f, s2 = 0.f;
        #pragma unroll
        for (int i = 0; i < 32; ++i) { float v = lp[lane + 32 * i]; xv[i] = v; s1 += v; s2 += v * v; }
        for (int off = 16; off > 0; off >>= 1) { s1 += __shfl_xor(s1, off, 32); s2 += __shfl_xor(s2, off, 32); }
        float mean = s1 * (1.f / DIM);
        float rstd = rsqrtf(s2 * (1.f / DIM) - mean * mean + 1e-5f);
        #pragma unroll
        for (int i = 0; i < 32; ++i) {
            int c = lane + 32 * i;
            lat[row * DIM + c] = (_Float16)((xv[i] - mean) * rstd * gl[c] + bl[c]);
        }
    }
    __syncthreads();

    const int arow   = lane & 15;
    const int ncol   = lane & 15;
    const int koffA  = (lane < 16) ? 0 : 16;
    const int koffB  = (lane < 16) ? 0 : 16;
    const int rowoff = (lane < 16) ? 0 : 8;
    const int rg     = wave >> 2;             // row-group (M rows rg*16..rg*16+15)
    const int nt4    = wave & 3;              // 16-col tile within the 64-wide head dim

    v8f outacc[2][8];
    #pragma unroll
    for (int g = 0; g < 2; ++g)
        #pragma unroll
        for (int u = 0; u < 8; ++u) outacc[g][u] = (v8f){};

    for (int h = 0; h < HEADS; ++h) {
        // --- q projection: 8 tiles (2 row-groups x 4 col-tiles), all 8 waves ---
        {
            v8f acc = {};
            const _Float16* ap = lat + (rg * 16 + arow) * DIM;
            const _Float16* wb = WqT + (size_t)(h * DHEAD + nt4 * 16 + ncol) * DIM;
            for (int kk = 0; kk < DIM; kk += 32) {
                v16h A = load_a(ap, kk + koffA);
                v16h B = *(const v16h*)(wb + kk + koffB);
                acc = wmma16(A, B, acc);
            }
            #pragma unroll
            for (int r = 0; r < 8; ++r)
                q16[(rg * 16 + r + rowoff) * 64 + nt4 * 16 + ncol] = (_Float16)(acc[r] * 0.125f);
        }
        __syncthreads();

        // --- sim = q @ k^T : K = DHEAD = 64, 8 tiles, all 8 waves ---
        {
            v8f acc = {};
            const _Float16* ap = q16 + (rg * 16 + arow) * 64;
            const _Float16* kb = K16 + (((size_t)b * HEADS + h) * SEQ + nt4 * 16 + ncol) * DHEAD;
            #pragma unroll
            for (int kk = 0; kk < DHEAD; kk += 32) {
                v16h A = load_a(ap, kk + koffA);
                v16h B = *(const v16h*)(kb + kk + koffB);
                acc = wmma16(A, B, acc);
            }
            #pragma unroll
            for (int r = 0; r < 8; ++r)
                simS[(rg * 16 + r + rowoff) * 64 + nt4 * 16 + ncol] = acc[r];
        }
        __syncthreads();

        // --- softmax over S=64 (32 rows x 8 threads, 8 cols each) ---
        {
            int row = threadIdx.x >> 3;
            int j0  = (threadIdx.x & 7) * 8;
            float e[8]; float m = -3.0e38f;
            #pragma unroll
            for (int j = 0; j < 8; ++j) { e[j] = simS[row * 64 + j0 + j]; m = fmaxf(m, e[j]); }
            for (int off = 4; off > 0; off >>= 1) m = fmaxf(m, __shfl_xor(m, off, 32));
            float s = 0.f;
            #pragma unroll
            for (int j = 0; j < 8; ++j) { e[j] = __expf(e[j] - m); s += e[j]; }
            for (int off = 4; off > 0; off >>= 1) s += __shfl_xor(s, off, 32);
            float inv = 1.0f / s;
            #pragma unroll
            for (int j = 0; j < 8; ++j) at16[row * 64 + j0 + j] = (_Float16)(e[j] * inv);
        }
        __syncthreads();

        // --- out_h = attn @ v^T (source quirk: contract attn's S with v's DH) ---
        {
            v8f acc = {};
            const _Float16* ap = at16 + (rg * 16 + arow) * 64;
            const _Float16* vb = V16 + (((size_t)b * HEADS + h) * SEQ + nt4 * 16 + ncol) * DHEAD;
            #pragma unroll
            for (int kk = 0; kk < 64; kk += 32) {
                v16h A = load_a(ap, kk + koffA);
                v16h B = *(const v16h*)(vb + kk + koffB);
                acc = wmma16(A, B, acc);
            }
            #pragma unroll
            for (int r = 0; r < 8; ++r)
                ao16[(rg * 16 + r + rowoff) * 64 + nt4 * 16 + ncol] = (_Float16)acc[r];
        }
        __syncthreads();

        // --- accumulate out-projection: outacc += ao16 @ Wout[h*64:(h+1)*64, :] ---
        // 128 tiles (2 row-groups x 64 col-tiles), 16 per wave
        #pragma unroll
        for (int g = 0; g < 2; ++g) {
            const _Float16* ap = ao16 + (g * 16 + arow) * 64;
            #pragma unroll
            for (int u = 0; u < 8; ++u) {
                int nrow = (wave + 8 * u) * 16 + ncol;
                const _Float16* wb = WoutT + (size_t)nrow * DIM + h * DHEAD;
                #pragma unroll
                for (int kk = 0; kk < 64; kk += 32) {
                    v16h A = load_a(ap, kk + koffA);
                    v16h B = *(const v16h*)(wb + kk + koffB);
                    outacc[g][u] = wmma16(A, B, outacc[g][u]);
                }
            }
        }
        __syncthreads();
    }

    // ---- store 32 x 1024 fp32 tile ----
    #pragma unroll
    for (int g = 0; g < 2; ++g) {
        #pragma unroll
        for (int u = 0; u < 8; ++u) {
            int n0 = (wave + 8 * u) * 16;
            #pragma unroll
            for (int r = 0; r < 8; ++r) {
                int row = m0 + g * 16 + r + rowoff;
                out[((size_t)b * MLAT + row) * DIM + n0 + ncol] = outacc[g][u][r];
            }
        }
    }
}

extern "C" void kernel_launch(void* const* d_in, const int* in_sizes, int n_in,
                              void* d_out, int out_size, void* d_ws, size_t ws_size,
                              hipStream_t stream) {
    const float* x       = (const float*)d_in[0];
    const float* latents = (const float*)d_in[1];
    const float* ln_x_g  = (const float*)d_in[2];
    const float* ln_x_b  = (const float*)d_in[3];
    const float* ln_l_g  = (const float*)d_in[4];
    const float* ln_l_b  = (const float*)d_in[5];
    const float* Wq      = (const float*)d_in[6];
    const float* Wkv     = (const float*)d_in[7];
    const float* Wout    = (const float*)d_in[8];
    float* out = (float*)d_out;

    // workspace layout (f16): WqT | WkvT | WoutT | K16 | V16  (12 MB total)
    _Float16* WqT   = (_Float16*)d_ws;
    _Float16* WkvT  = WqT  + (size_t)DIM * DIM;
    _Float16* WoutT = WkvT + (size_t)DIM * 2 * DIM;
    _Float16* K16   = WoutT + (size_t)DIM * DIM;
    _Float16* V16   = K16  + (size_t)BATCH * HEADS * SEQ * DHEAD;

    // K0: weight transpose + f16 convert (4M elements)
    {
        int total = DIM * DIM + DIM * 2 * DIM + DIM * DIM;
        wcvt_kernel<<<(total + 255) / 256, 256, 0, stream>>>(Wq, Wkv, Wout, WqT, WkvT, WoutT);
    }
    // K1: LN(x) + kv projection
    kv_kernel<<<BATCH * (SEQ / 16), 256, 0, stream>>>(x, ln_x_g, ln_x_b, WkvT, K16, V16);
    // K2: fused LN(latents) + q + attention + out projection (84 KB dynamic LDS)
    attn_kernel<<<BATCH * (MLAT / 32), 256, ATTN_LDS_BYTES, stream>>>(latents, ln_l_g, ln_l_b,
                                                                      WqT, WoutT, K16, V16, out);
}
